// RBYTransformer_62801011802983
// MI455X (gfx1250) — compile-verified, matrix operand order checked by reading, not run on hardware
//
#include <hip/hip_runtime.h>
#include <hip/hip_fp16.h>

typedef __attribute__((ext_vector_type(16))) _Float16 v16h;
typedef __attribute__((ext_vector_type(8)))  _Float16 v8h;
typedef __attribute__((ext_vector_type(8)))  float    v8f;

#define D_MODEL 1024
#define SEQ     2048
#define BATCH   4
#define NHEADS  16
#define NTOK    (BATCH * SEQ)          // 8192 rows
#define LDSP    40                     // padded LDS row stride in halves (80B)

union AF { struct { uint4 lo, hi; } u; v16h h; };

// ---------------------------------------------------------------------------
// adaptive positional encoding: phase_mod = 0.1*sin(2*pi*0.25) = 0.1
// ---------------------------------------------------------------------------
static __device__ __forceinline__ float pos_enc(int s, int n, float yellow) {
    int ie = n & ~1;
    float denom = __powf(10000.0f, (2.0f * (float)ie) * (1.0f / 1024.0f)) *
                  (1.0f + yellow * 0.1f);
    float ang = (float)s / denom;
    return (n & 1) ? __cosf(ang) : __sinf(ang);
}

__global__ void prep_x(const float* __restrict__ x, const float* __restrict__ rby,
                       _Float16* __restrict__ xh) {
    int idx = blockIdx.x * 256 + threadIdx.x;
    int n = idx & (D_MODEL - 1);
    int s = (idx >> 10) & (SEQ - 1);
    float yellow = rby[2];
    xh[idx] = (_Float16)(x[idx] + pos_enc(s, n, yellow));
}

__global__ void cvt_h(const float* __restrict__ src, _Float16* __restrict__ dst, int n) {
    int i = blockIdx.x * 256 + threadIdx.x;
    if (i < n) dst[i] = (_Float16)src[i];
}

// ---------------------------------------------------------------------------
// Tiled f16 GEMM with double-buffered ASYNC global->LDS tile staging.
// C[M,N] = A[M,K] * B[N,K]^T (+bias [+residual+pe for MODE1])
// block = 128 threads (4 waves), 64x64 tile, each wave 32x32 (2x2 WMMA tiles)
// ---------------------------------------------------------------------------
template <int MODE>
__global__ __launch_bounds__(128)
void gemm64(const _Float16* __restrict__ A, const _Float16* __restrict__ B,
            const float* __restrict__ bias,
            _Float16* __restrict__ outH, float* __restrict__ outF,
            const float* __restrict__ xres, const float* __restrict__ rby,
            int M, int N, int K) {
    __shared__ __align__(16) _Float16 sA[2][64 * LDSP];
    __shared__ __align__(16) _Float16 sB[2][64 * LDSP];

    const int tid  = threadIdx.x;
    const int lane = tid & 31;
    const int wave = tid >> 5;
    const int lr   = lane & 15;
    const int hf   = lane >> 4;
    const int wm   = (wave >> 1) * 32;
    const int wn   = (wave & 1) * 32;
    const int m0   = blockIdx.y * 64;
    const int n0   = blockIdx.x * 64;

    const int ldRow = tid >> 1;          // 0..63
    const int ldCol = (tid & 1) * 16;    // 0 or 16 halves

    // async copy of this thread's 32B slice of the A and B tiles (4 ops/thread)
    auto issue_tile = [&](int buf, int k0) {
        unsigned long long ga =
            (unsigned long long)(uintptr_t)(A + (size_t)(m0 + ldRow) * K + k0 + ldCol);
        unsigned long long gb =
            (unsigned long long)(uintptr_t)(B + (size_t)(n0 + ldRow) * K + k0 + ldCol);
        unsigned la = (unsigned)(uintptr_t)(sA[buf] + ldRow * LDSP + ldCol);
        unsigned lb = (unsigned)(uintptr_t)(sB[buf] + ldRow * LDSP + ldCol);
        // INST_OFFSET applies to both the global and LDS sides (ISA 15.18.3)
        asm volatile("global_load_async_to_lds_b128 %0, %1, off"
                     :: "v"(la), "v"(ga) : "memory");
        asm volatile("global_load_async_to_lds_b128 %0, %1, off offset:16"
                     :: "v"(la), "v"(ga) : "memory");
        asm volatile("global_load_async_to_lds_b128 %0, %1, off"
                     :: "v"(lb), "v"(gb) : "memory");
        asm volatile("global_load_async_to_lds_b128 %0, %1, off offset:16"
                     :: "v"(lb), "v"(gb) : "memory");
    };

    v8f acc[2][2] = {};

    issue_tile(0, 0);
    for (int k0 = 0, it = 0; k0 < K; k0 += 32, ++it) {
        const int cur = it & 1;
        if (k0 + 32 < K) {
            issue_tile(cur ^ 1, k0 + 32);
            // previous tile (4 ops) done; prefetch (4 ops) may remain in flight
            asm volatile("s_wait_asynccnt 0x4" ::: "memory");
        } else {
            asm volatile("s_wait_asynccnt 0x0" ::: "memory");
        }
        __syncthreads();

        v16h af[2], bf[2];
#pragma unroll
        for (int t = 0; t < 2; t++) {
            const _Float16* ap = sA[cur] + (wm + t * 16 + lr) * LDSP;
            AF fa;
            fa.u.lo = *(const uint4*)(ap + hf * 8);
            fa.u.hi = *(const uint4*)(ap + 16 + hf * 8);
            af[t] = fa.h;
            const _Float16* bp = sB[cur] + (wn + t * 16 + lr) * LDSP;
            AF fb;
            fb.u.lo = *(const uint4*)(bp + hf * 8);
            fb.u.hi = *(const uint4*)(bp + 16 + hf * 8);
            bf[t] = fb.h;
        }
#pragma unroll
        for (int tm = 0; tm < 2; tm++)
#pragma unroll
            for (int tn = 0; tn < 2; tn++)
                acc[tm][tn] = __builtin_amdgcn_wmma_f32_16x16x32_f16(
                    false, af[tm], false, bf[tn], (short)0, acc[tm][tn], false, false);
        __syncthreads();
    }

    float yellow = rby[2];
#pragma unroll
    for (int tm = 0; tm < 2; tm++)
#pragma unroll
        for (int tn = 0; tn < 2; tn++)
#pragma unroll
            for (int v = 0; v < 8; v++) {
                int m = m0 + wm + tm * 16 + v + 8 * hf;
                int n = n0 + wn + tn * 16 + lr;
                float val = acc[tm][tn][v] + bias[n];
                if (MODE == 0) {
                    outH[(size_t)m * N + n] = (_Float16)val;
                } else {
                    int s_ = m & (SEQ - 1);
                    float pe = pos_enc(s_, n, yellow);
                    outF[(size_t)m * N + n] = val + xres[(size_t)m * N + n] + pe;
                }
            }
}

// ---------------------------------------------------------------------------
// Flash attention, S^T formulation: compute S^T = K * Q^T so keys land in the
// accumulator VGPR index and queries in lanes -> softmax reductions over keys
// are pure VALU + one shfl_xor(16); per-lane state is just (m, l).
// One block = (b, h, 64 q-rows); 4 waves x 16 q-rows.
// ---------------------------------------------------------------------------
__global__ __launch_bounds__(128)
void attn_kernel(const _Float16* __restrict__ Qh, const _Float16* __restrict__ Kh,
                 const _Float16* __restrict__ Vh, const float* __restrict__ rby,
                 _Float16* __restrict__ Oh) {
    __shared__ __align__(16) _Float16 sV[64 * LDSP];     // V^T tile: [d=64][kk=32]
    __shared__ __align__(16) _Float16 sP[4][16 * LDSP];  // per-wave P tile [q=16][kk=32]

    const int tid  = threadIdx.x;
    const int lane = tid & 31;
    const int wave = tid >> 5;
    const int lr   = lane & 15;
    const int hf   = lane >> 4;

    const int bid = blockIdx.x;          // B*H*(S/64) = 2048
    const int qb  = bid & 31;
    const int h   = (bid >> 5) & 15;
    const int b   = bid >> 9;
    const int q0  = qb * 64 + wave * 16;
    const size_t rowBase = (size_t)b * SEQ;

    const float red = rby[0], blue = rby[1], yellow = rby[2];
    // logits scale: scale_matrix / sqrt(64)  (TENSION == 0)
    const float lscale = (1.0f + red * 0.1f + blue * 0.05f + yellow * 0.15f) * 0.125f;
    const float oscale = 1.0f + yellow * 1.2246468e-17f;  // 1 + y*0.1*sin(pi)

    // Q fragments: rows are [query][d] = exactly the N x K major B-operand
    v16h qf[2];
    {
        const _Float16* qp = Qh + (rowBase + q0 + lr) * D_MODEL + h * 64;
#pragma unroll
        for (int d2 = 0; d2 < 2; d2++) {
            AF f;
            f.u.lo = *(const uint4*)(qp + d2 * 32 + hf * 8);
            f.u.hi = *(const uint4*)(qp + d2 * 32 + 16 + hf * 8);
            qf[d2] = f.h;
        }
    }

    // per-lane streaming-softmax state for query q0 + lr (replicated over hf)
    float mst = -3.0e38f, lst = 0.0f;
    v8f acc[4] = {};

    const int vk = tid & 31;   // key index for V staging
    const int vd = tid >> 5;   // d-quarter 0..3

    for (int j = 0; j < SEQ; j += 32) {
        // --- stage V tile transposed into LDS: sV[d][kk] ---
        const _Float16* vp = Vh + (rowBase + j + vk) * D_MODEL + h * 64 + vd * 16;
        v8h v0 = *(const v8h*)vp;
        v8h v1 = *(const v8h*)(vp + 8);
#pragma unroll
        for (int e = 0; e < 8; e++) {
            sV[(vd * 16 + e) * LDSP + vk]     = v0[e];
            sV[(vd * 16 + 8 + e) * LDSP + vk] = v1[e];
        }
        __syncthreads();

        // --- S^T = K(16x64) * Q^T : A = K rows, B = Q rows ---
        v8f st[2];
#pragma unroll
        for (int ct = 0; ct < 2; ct++) {
            const _Float16* kp = Kh + (rowBase + j + ct * 16 + lr) * D_MODEL + h * 64;
            AF k0, k1;
            k0.u.lo = *(const uint4*)(kp + hf * 8);
            k0.u.hi = *(const uint4*)(kp + 16 + hf * 8);
            k1.u.lo = *(const uint4*)(kp + 32 + hf * 8);
            k1.u.hi = *(const uint4*)(kp + 48 + hf * 8);
            v8f z = {0.f, 0.f, 0.f, 0.f, 0.f, 0.f, 0.f, 0.f};
            z = __builtin_amdgcn_wmma_f32_16x16x32_f16(false, k0.h, false, qf[0],
                                                       (short)0, z, false, false);
            st[ct] = __builtin_amdgcn_wmma_f32_16x16x32_f16(false, k1.h, false, qf[1],
                                                            (short)0, z, false, false);
#pragma unroll
            for (int v = 0; v < 8; v++) st[ct][v] *= lscale;
        }

        // --- column (query) max over 32 keys: VALU over regs + one xor(16) ---
        float t = fmaxf(st[0][0], st[1][0]);
#pragma unroll
        for (int v = 1; v < 8; v++) t = fmaxf(t, fmaxf(st[0][v], st[1][v]));
        t = fmaxf(t, __shfl_xor(t, 16));
        float mn = fmaxf(mst, t);

        // --- exp + column sum; write P^T into per-wave LDS as P[q][kk] ---
        _Float16* pw = sP[wave];
        float rs = 0.0f;
#pragma unroll
        for (int ct = 0; ct < 2; ct++)
#pragma unroll
            for (int v = 0; v < 8; v++) {
                float p = __expf(st[ct][v] - mn);
                rs += p;
                pw[lr * LDSP + ct * 16 + v + 8 * hf] = (_Float16)p;
            }
        rs += __shfl_xor(rs, 16);
        float corr = __expf(mst - mn);
        lst = lst * corr + rs;
        mst = mn;

        // --- rescale acc: out tile is query-major -> broadcast corr per row ---
#pragma unroll
        for (int v = 0; v < 8; v++) {
            float cv = __shfl(corr, v + 8 * hf, 32);
#pragma unroll
            for (int nt = 0; nt < 4; nt++) acc[nt][v] *= cv;
        }

        // --- P fragment (A layout) from LDS bounce ---
        AF pf;
        pf.u.lo = *(const uint4*)(pw + lr * LDSP + hf * 8);
        pf.u.hi = *(const uint4*)(pw + lr * LDSP + 16 + hf * 8);

        // --- acc += P(16x32) * V(32x64) ---
#pragma unroll
        for (int nt = 0; nt < 4; nt++) {
            const _Float16* vb = sV + (nt * 16 + lr) * LDSP;
            AF vf;
            vf.u.lo = *(const uint4*)(vb + hf * 8);
            vf.u.hi = *(const uint4*)(vb + 16 + hf * 8);
            acc[nt] = __builtin_amdgcn_wmma_f32_16x16x32_f16(false, pf.h, false, vf.h,
                                                             (short)0, acc[nt], false, false);
        }
        __syncthreads();
    }

    // --- epilogue: broadcast 1/l per query row, store f16 ---
    float invl = oscale / lst;
#pragma unroll
    for (int v = 0; v < 8; v++) {
        float iv = __shfl(invl, v + 8 * hf, 32);
        int m = q0 + v + 8 * hf;
        _Float16* op = Oh + (rowBase + m) * D_MODEL + h * 64 + lr;
#pragma unroll
        for (int nt = 0; nt < 4; nt++)
            op[nt * 16] = (_Float16)(acc[nt][v] * iv);
    }
}

// ---------------------------------------------------------------------------
// RBY-conditioned LayerNorm: one row per block, 256 threads x 4 cols
// ---------------------------------------------------------------------------
__global__ void ln_kernel(const float* __restrict__ y, const float* __restrict__ rby,
                          const float* __restrict__ gamma, const float* __restrict__ beta,
                          float* __restrict__ out) {
    __shared__ float ws1[8], ws2[8];
    const int row = blockIdx.x;
    const int tid = threadIdx.x;
    const float* yr = y + (size_t)row * D_MODEL;
    float4 val = *(const float4*)(yr + tid * 4);
    float s  = val.x + val.y + val.z + val.w;
    float s2 = val.x * val.x + val.y * val.y + val.z * val.z + val.w * val.w;
#pragma unroll
    for (int mask = 1; mask < 32; mask <<= 1) {
        s  += __shfl_xor(s, mask);
        s2 += __shfl_xor(s2, mask);
    }
    if ((tid & 31) == 0) { ws1[tid >> 5] = s; ws2[tid >> 5] = s2; }
    __syncthreads();
    float S = 0.f, S2 = 0.f;
#pragma unroll
    for (int i = 0; i < 8; i++) { S += ws1[i]; S2 += ws2[i]; }
    float mean = S * (1.0f / D_MODEL);
    float var  = S2 * (1.0f / D_MODEL) - mean * mean;
    float red = rby[0], blue = rby[1], yellow = rby[2];
    float eps  = 1e-5f * (1.0f + blue * 0.1f);
    float gs   = 1.0f + red * 0.1f + yellow * 0.05f;
    float rstd = rsqrtf(var + eps);
    int n = tid * 4;
    float* o = out + (size_t)row * D_MODEL + n;
    o[0] = (val.x - mean) * rstd * gamma[n + 0] * gs + beta[n + 0];
    o[1] = (val.y - mean) * rstd * gamma[n + 1] * gs + beta[n + 1];
    o[2] = (val.z - mean) * rstd * gamma[n + 2] * gs + beta[n + 2];
    o[3] = (val.w - mean) * rstd * gamma[n + 3] * gs + beta[n + 3];
}

// ---------------------------------------------------------------------------
extern "C" void kernel_launch(void* const* d_in, const int* in_sizes, int n_in,
                              void* d_out, int out_size, void* d_ws, size_t ws_size,
                              hipStream_t stream) {
    const float* x   = (const float*)d_in[0];
    const float* rby = (const float*)d_in[1];
    const float* Wq  = (const float*)d_in[2];
    const float* bq  = (const float*)d_in[3];
    const float* Wk  = (const float*)d_in[4];
    const float* bk  = (const float*)d_in[5];
    const float* Wv  = (const float*)d_in[6];
    const float* bv  = (const float*)d_in[7];
    const float* Wo  = (const float*)d_in[8];
    const float* bo  = (const float*)d_in[9];
    const float* g   = (const float*)d_in[10];
    const float* be  = (const float*)d_in[11];
    float* out = (float*)d_out;

    char* ws = (char*)d_ws;
    const size_t tokBytesH = (size_t)NTOK * D_MODEL * sizeof(_Float16);
    const size_t wBytesH   = (size_t)D_MODEL * D_MODEL * sizeof(_Float16);
    _Float16* xh  = (_Float16*)ws;  ws += tokBytesH;
    _Float16* wqh = (_Float16*)ws;  ws += wBytesH;
    _Float16* wkh = (_Float16*)ws;  ws += wBytesH;
    _Float16* wvh = (_Float16*)ws;  ws += wBytesH;
    _Float16* woh = (_Float16*)ws;  ws += wBytesH;
    _Float16* qh  = (_Float16*)ws;  ws += tokBytesH;
    _Float16* kh  = (_Float16*)ws;  ws += tokBytesH;
    _Float16* vh  = (_Float16*)ws;  ws += tokBytesH;
    _Float16* oh  = (_Float16*)ws;  ws += tokBytesH;
    float*    y   = (float*)ws;     ws += (size_t)NTOK * D_MODEL * sizeof(float);

    prep_x<<<(NTOK * D_MODEL) / 256, 256, 0, stream>>>(x, rby, xh);
    const int welems = D_MODEL * D_MODEL;
    cvt_h<<<welems / 256, 256, 0, stream>>>(Wq, wqh, welems);
    cvt_h<<<welems / 256, 256, 0, stream>>>(Wk, wkh, welems);
    cvt_h<<<welems / 256, 256, 0, stream>>>(Wv, wvh, welems);
    cvt_h<<<welems / 256, 256, 0, stream>>>(Wo, woh, welems);

    dim3 gg(D_MODEL / 64, NTOK / 64);   // (16, 128)
    gemm64<0><<<gg, 128, 0, stream>>>(xh, wqh, bq, qh, nullptr, nullptr, rby,
                                      NTOK, D_MODEL, D_MODEL);
    gemm64<0><<<gg, 128, 0, stream>>>(xh, wkh, bk, kh, nullptr, nullptr, rby,
                                      NTOK, D_MODEL, D_MODEL);
    gemm64<0><<<gg, 128, 0, stream>>>(xh, wvh, bv, vh, nullptr, nullptr, rby,
                                      NTOK, D_MODEL, D_MODEL);

    attn_kernel<<<BATCH * NHEADS * (SEQ / 64), 128, 0, stream>>>(qh, kh, vh, rby, oh);

    gemm64<1><<<gg, 128, 0, stream>>>(oh, woh, bo, nullptr, y, x, rby,
                                      NTOK, D_MODEL, D_MODEL);

    ln_kernel<<<NTOK, 256, 0, stream>>>(y, rby, g, be, out);
}
